// CCRGNN_16621523436376
// MI455X (gfx1250) — compile-verified, matrix operand order checked by reading, not run on hardware
//
#include <hip/hip_runtime.h>
#include <hip/hip_bf16.h>

#define BGRAPHS 8192
#define NPG     39
#define NTOT    (BGRAPHS * NPG)      // 319488 nodes
#define EPG     (NPG * 8)            // 312 edges per graph
#define ETOT    ((size_t)BGRAPHS * EPG)
#define FCOLS   4560
#define FP      4576                 // padded to multiple of 32 for WMMA K

typedef __bf16 bf16_t;
typedef bf16_t v8bf  __attribute__((ext_vector_type(8)));
typedef bf16_t v16bf __attribute__((ext_vector_type(16)));
typedef float  v8f   __attribute__((ext_vector_type(8)));

// ---------------------------------------------------------------------------
// Convert f32 weight [K,N] row-major -> bf16 transposed [N,Kp] (n-major, K pad 0)
// ---------------------------------------------------------------------------
__global__ void convT_kernel(const float* __restrict__ w, bf16_t* __restrict__ wt,
                             int K, int N, int Kp) {
    int idx = blockIdx.x * 256 + threadIdx.x;
    if (idx >= N * Kp) return;
    int n = idx / Kp;
    int k = idx - n * Kp;
    wt[idx] = (k < K) ? (bf16_t)w[(size_t)k * N + n] : (bf16_t)0.f;
}

// ---------------------------------------------------------------------------
// Raw-input residual + pool(x) + zero the K-pad columns of f
// ---------------------------------------------------------------------------
__global__ void xres_kernel(const float* __restrict__ x, bf16_t* __restrict__ f) {
    int g = blockIdx.x, tid = threadIdx.x;
    __shared__ float xs[NPG];
    if (tid < NPG) {
        float v = x[(size_t)g * NPG + tid];
        xs[tid] = v;
        f[(size_t)g * FP + tid] = (bf16_t)v;           // res: cols 0..38
    }
    if (tid < FP - FCOLS)
        f[(size_t)g * FP + FCOLS + tid] = (bf16_t)0.f; // K padding
    __syncthreads();
    if (tid == 0) {
        float m = xs[0];
        for (int i = 1; i < NPG; i++) m = fmaxf(m, xs[i]);
        f[(size_t)g * FP + 4446] = (bf16_t)m;          // out0
    }
}

// ---------------------------------------------------------------------------
// One GAT layer, one workgroup per graph, everything in LDS.
//  h = x@W ; e_ij = leaky(a_s.h[src] + a_d.h[dst]) ; softmax per dst (incl.
//  implicit self-loop, since reference appends self-loops) ; aggregate ; +b ;
//  relu ; write h_out, residual slice of f (bf16) and pooled max (bf16).
// ---------------------------------------------------------------------------
template <int FI, int FO>
__launch_bounds__(256)
__global__ void gat_kernel(const float* __restrict__ xin,
                           const long long* __restrict__ src_g,
                           const long long* __restrict__ dst_g,
                           const float* __restrict__ W,
                           const float* __restrict__ a_s,
                           const float* __restrict__ a_d,
                           const float* __restrict__ bias,
                           float* __restrict__ hout,
                           bf16_t* __restrict__ f,
                           int res_off, int out_off) {
    __shared__ float sW[FI * FO];
    __shared__ float sas[FO], sad[FO], sb[FO];
    __shared__ float xl[NPG * FI];
    __shared__ float h[NPG * FO];
    __shared__ float ho[NPG * FO];
    __shared__ float sn[NPG], dn[NPG];
    __shared__ unsigned short srcl[EPG], dstl[EPG];
    __shared__ float ev[EPG];
    __shared__ float slw[NPG], mx[NPG], den[NPG];
    __shared__ int cnt[NPG], off[NPG + 1], cur[NPG];
    __shared__ unsigned short perm[EPG];

    const int g = blockIdx.x;
    const int tid = threadIdx.x;
    const long long nbase = (long long)g * NPG;
    const size_t ebase = (size_t)g * EPG;

    for (int i = tid; i < FI * FO; i += 256) sW[i] = W[i];
    for (int i = tid; i < FO; i += 256) { sas[i] = a_s[i]; sad[i] = a_d[i]; sb[i] = bias[i]; }
    for (int i = tid; i < NPG * FI; i += 256) xl[i] = xin[(size_t)nbase * FI + i];
    for (int e = tid; e < EPG; e += 256) {
        srcl[e] = (unsigned short)(src_g[ebase + e] - nbase);
        dstl[e] = (unsigned short)(dst_g[ebase + e] - nbase);
    }
    if (tid < NPG) cnt[tid] = 0;
    __syncthreads();

    // h = x @ W
    for (int p = tid; p < NPG * FO; p += 256) {
        int n = p / FO, fo = p - n * FO;
        float acc = 0.f;
        #pragma unroll
        for (int i = 0; i < FI; i++) acc += xl[n * FI + i] * sW[i * FO + fo];
        h[p] = acc;
    }
    // CSR bucket edges by dst (LDS atomics)
    for (int e = tid; e < EPG; e += 256) atomicAdd(&cnt[dstl[e]], 1);
    __syncthreads();
    if (tid < NPG) {
        float a = 0.f, b = 0.f;
        #pragma unroll
        for (int fo = 0; fo < FO; fo++) { float hv = h[tid * FO + fo]; a += hv * sas[fo]; b += hv * sad[fo]; }
        sn[tid] = a; dn[tid] = b;
    }
    if (tid == 0) {
        int o = 0;
        for (int t = 0; t < NPG; t++) { off[t] = o; o += cnt[t]; }
        off[NPG] = o;
    }
    __syncthreads();
    if (tid < NPG) cur[tid] = off[tid];
    __syncthreads();
    for (int e = tid; e < EPG; e += 256) {
        int p = atomicAdd(&cur[dstl[e]], 1);
        perm[p] = (unsigned short)e;
    }
    __syncthreads();
    // edge logits (leaky relu, slope 0.2)
    for (int e = tid; e < EPG; e += 256) {
        float v = sn[srcl[e]] + dn[dstl[e]];
        ev[e] = v > 0.f ? v : 0.2f * v;
    }
    __syncthreads();
    // per-dst max & denom (self-loop included implicitly)
    if (tid < NPG) {
        float sv = sn[tid] + dn[tid];
        sv = sv > 0.f ? sv : 0.2f * sv;
        float m = sv;
        for (int j = off[tid]; j < off[tid + 1]; j++) m = fmaxf(m, ev[perm[j]]);
        mx[tid] = m;
        float sw = __expf(sv - m);
        slw[tid] = sw;
        float d = sw;
        for (int j = off[tid]; j < off[tid + 1]; j++) d += __expf(ev[perm[j]] - m);
        den[tid] = d;
    }
    __syncthreads();
    // edge weights (overwrite ev)
    for (int e = tid; e < EPG; e += 256) ev[e] = __expf(ev[e] - mx[dstl[e]]);
    __syncthreads();
    // aggregate, bias, relu, write out + residual slice of f
    for (int p = tid; p < NPG * FO; p += 256) {
        int t = p / FO, fo = p - t * FO;
        float acc = slw[t] * h[t * FO + fo];
        for (int j = off[t]; j < off[t + 1]; j++) {
            int e = perm[j];
            acc += ev[e] * h[srcl[e] * FO + fo];
        }
        float o = acc / (den[t] + 1e-16f) + sb[fo];
        o = fmaxf(o, 0.f);
        ho[p] = o;
        hout[(size_t)(nbase + t) * FO + fo] = o;
        f[(size_t)g * FP + res_off + t * FO + fo] = (bf16_t)o;
    }
    __syncthreads();
    // pooled max per feature
    if (tid < FO) {
        float m = ho[tid];
        for (int t = 1; t < NPG; t++) m = fmaxf(m, ho[t * FO + tid]);
        f[(size_t)g * FP + out_off + tid] = (bf16_t)m;
    }
}

// ---------------------------------------------------------------------------
// bf16 WMMA GEMM: C[M,N] = relu(A[M,Kp] @ B[Kp,N] + bias)
// A row-major bf16 (lda=Kp), B supplied transposed n-major [N][Kp] bf16.
// Block: 256 threads = 8 waves; tile 64(M) x 128(N); K-step 32.
// Each wave computes a 32x32 sub-tile as 2x2 v_wmma_f32_16x16x32_bf16.
// ---------------------------------------------------------------------------
__launch_bounds__(256)
__global__ void gemm_wmma_bf16(const bf16_t* __restrict__ A, int lda,
                               const bf16_t* __restrict__ Bt, int ldb,
                               const float* __restrict__ bias,
                               bf16_t* __restrict__ outB, float* __restrict__ outF,
                               int N_total, int Kp, int relu) {
    __shared__ __align__(16) bf16_t Al[64][40];    // +8 pad, rows 80B (16B aligned)
    __shared__ __align__(16) bf16_t Bl[128][40];

    const int tid = threadIdx.x;
    const int mbase = blockIdx.x * 64;
    const int nbase = blockIdx.y * 128;
    const int wid = tid >> 5, lane = tid & 31;
    const int wm = (wid >> 2) * 32;     // 0 / 32
    const int wn = (wid & 3) * 32;      // 0..96
    const int lr = lane & 15;
    const int kb = (lane >> 4) * 8;     // K half select for 16-bit A/B layout

    v8f acc[2][2];
    #pragma unroll
    for (int i = 0; i < 2; i++)
        #pragma unroll
        for (int j = 0; j < 2; j++)
            #pragma unroll
            for (int e = 0; e < 8; e++) acc[i][j][e] = 0.f;

    const int ksteps = Kp >> 5;
    for (int kt = 0; kt < ksteps; ++kt) {
        const int k0 = kt << 5;
        {   // A tile: 64 rows x 32 cols, one 16B chunk per thread
            int row = tid >> 2, ch = (tid & 3) << 3;
            *(v8bf*)&Al[row][ch] =
                *(const v8bf*)&A[(size_t)(mbase + row) * lda + k0 + ch];
        }
        #pragma unroll
        for (int j = 0; j < 2; j++) {   // B tile: 128 rows x 32 cols
            int idx = tid + (j << 8);
            int row = idx >> 2, ch = (idx & 3) << 3;
            *(v8bf*)&Bl[row][ch] =
                *(const v8bf*)&Bt[(size_t)(nbase + row) * ldb + k0 + ch];
        }
        __syncthreads();

        // Fragments per ISA 16-bit A/B striping: lane holds row (lane&15),
        // K = kb..kb+7 in regs 0..3 and kb+16..kb+23 in regs 4..7.
        union frag { v16bf v; v8bf h[2]; };
        frag af[2], bfr[2];
        #pragma unroll
        for (int i = 0; i < 2; i++) {
            int r = wm + i * 16 + lr;
            af[i].h[0] = *(const v8bf*)&Al[r][kb];
            af[i].h[1] = *(const v8bf*)&Al[r][kb + 16];
            int c = wn + i * 16 + lr;
            bfr[i].h[0] = *(const v8bf*)&Bl[c][kb];
            bfr[i].h[1] = *(const v8bf*)&Bl[c][kb + 16];
        }
        #pragma unroll
        for (int i = 0; i < 2; i++)
            #pragma unroll
            for (int j = 0; j < 2; j++)
                acc[i][j] = __builtin_amdgcn_wmma_f32_16x16x32_bf16(
                    false, af[i].v, false, bfr[j].v, (short)0, acc[i][j],
                    false, false);
        __syncthreads();
    }

    // Epilogue: C/D layout — reg e: M = e + 8*(lane>>4), N = lane&15.
    const int mh = (lane >> 4) * 8;
    #pragma unroll
    for (int i = 0; i < 2; i++) {
        #pragma unroll
        for (int j = 0; j < 2; j++) {
            int n = nbase + wn + j * 16 + lr;
            float bv = bias[n];
            #pragma unroll
            for (int e = 0; e < 8; e++) {
                int m = mbase + wm + i * 16 + mh + e;
                float v = acc[i][j][e] + bv;
                if (relu) v = fmaxf(v, 0.f);
                if (outB) outB[(size_t)m * N_total + n] = (bf16_t)v;
                if (outF) outF[(size_t)m * N_total + n] = v;
            }
        }
    }
}

// ---------------------------------------------------------------------------
// Final tiny GEMM: out[B,9] = act2[B,128] @ lw3[128,9] + lb3 (no relu, f32)
// ---------------------------------------------------------------------------
__global__ void mlp3_kernel(const float* __restrict__ a, const float* __restrict__ w,
                            const float* __restrict__ b, float* __restrict__ out) {
    __shared__ float sw[128 * 9];
    __shared__ float sb2[9];
    int tid = threadIdx.x;
    for (int i = tid; i < 128 * 9; i += 256) sw[i] = w[i];
    if (tid < 9) sb2[tid] = b[tid];
    __syncthreads();
    int idx = blockIdx.x * 256 + tid;
    if (idx >= BGRAPHS * 9) return;
    int g = idx / 9, j = idx - g * 9;
    float acc = sb2[j];
    const float* ar = a + (size_t)g * 128;
    #pragma unroll 8
    for (int k = 0; k < 128; k++) acc += ar[k] * sw[k * 9 + j];
    out[idx] = acc;
}

// ---------------------------------------------------------------------------
extern "C" void kernel_launch(void* const* d_in, const int* in_sizes, int n_in,
                              void* d_out, int out_size, void* d_ws, size_t ws_size,
                              hipStream_t stream) {
    const float* x = (const float*)d_in[0];
    const long long* ei = (const long long*)d_in[1];
    // d_in[2] batch_ids unused: graph membership is structural (39 nodes/graph)
    const float* W1 = (const float*)d_in[3],  *as1 = (const float*)d_in[4],
               *ad1 = (const float*)d_in[5],  *b1  = (const float*)d_in[6];
    const float* W2 = (const float*)d_in[7],  *as2 = (const float*)d_in[8],
               *ad2 = (const float*)d_in[9],  *b2  = (const float*)d_in[10];
    const float* W3 = (const float*)d_in[11], *as3 = (const float*)d_in[12],
               *ad3 = (const float*)d_in[13], *b3  = (const float*)d_in[14];
    const float* W4 = (const float*)d_in[15], *as4 = (const float*)d_in[16],
               *ad4 = (const float*)d_in[17], *b4  = (const float*)d_in[18];
    const float* lw1 = (const float*)d_in[19], *lb1 = (const float*)d_in[20];
    const float* lw2 = (const float*)d_in[21], *lb2 = (const float*)d_in[22];
    const float* lw3 = (const float*)d_in[23], *lb3 = (const float*)d_in[24];
    float* out = (float*)d_out;

    const long long* src_g = ei;
    const long long* dst_g = ei + ETOT;

    // workspace layout (all 256B aligned by construction)
    char* ws = (char*)d_ws;
    size_t o = 0;
    bf16_t* f    = (bf16_t*)(ws + o); o += (size_t)BGRAPHS * FP * 2;   // 74.97 MB
    float*  hA   = (float*)(ws + o);  o += (size_t)NTOT * 32 * 4;      // 40.89 MB (h1, h3)
    float*  hB   = (float*)(ws + o);  o += (size_t)NTOT * 64 * 4;      // 81.79 MB (h2, h4)
    bf16_t* lw1T = (bf16_t*)(ws + o); o += (size_t)1024 * FP * 2;      //  9.37 MB
    bf16_t* lw2T = (bf16_t*)(ws + o); o += (size_t)128 * 1024 * 2;     //  0.26 MB
    // aliases: GAT feature buffers are dead once f is built
    bf16_t* act1 = (bf16_t*)hA;                                        // 16.78 MB
    float*  act2 = (float*)((char*)hA + (size_t)BGRAPHS * 1024 * 2);   //  4.19 MB

    // 1. transpose+convert MLP weights to bf16 (K padded with zeros)
    convT_kernel<<<(1024 * FP + 255) / 256, 256, 0, stream>>>(lw1, lw1T, FCOLS, 1024, FP);
    convT_kernel<<<(128 * 1024 + 255) / 256, 256, 0, stream>>>(lw2, lw2T, 1024, 128, 1024);

    // 2. raw residual + pool(x) + zero f's pad columns
    xres_kernel<<<BGRAPHS, 64, 0, stream>>>(x, f);

    // 3. four GAT layers, one workgroup per graph, all-LDS
    gat_kernel<1, 8><<<BGRAPHS, 256, 0, stream>>>(x,  src_g, dst_g, W1, as1, ad1, b1, hA, f,   39, 4447);
    gat_kernel<8, 64><<<BGRAPHS, 256, 0, stream>>>(hA, src_g, dst_g, W2, as2, ad2, b2, hB, f,  351, 4455);
    gat_kernel<64, 32><<<BGRAPHS, 256, 0, stream>>>(hB, src_g, dst_g, W3, as3, ad3, b3, hA, f, 2847, 4519);
    gat_kernel<32, 9><<<BGRAPHS, 256, 0, stream>>>(hA, src_g, dst_g, W4, as4, ad4, b4, hB, f, 4095, 4551);

    // 4. MLP1: [8192,4576]bf16 @ lw1T -> relu -> act1 bf16 [8192,1024]
    gemm_wmma_bf16<<<dim3(BGRAPHS / 64, 1024 / 128), 256, 0, stream>>>(
        f, FP, lw1T, FP, lb1, act1, nullptr, 1024, FP, 1);

    // 5. MLP2: [8192,1024]bf16 @ lw2T -> relu -> act2 f32 [8192,128]
    gemm_wmma_bf16<<<dim3(BGRAPHS / 64, 1), 256, 0, stream>>>(
        act1, 1024, lw2T, 1024, lb2, nullptr, act2, 128, 1024, 1);

    // 6. MLP3 -> d_out [8192,9] f32
    mlp3_kernel<<<(BGRAPHS * 9 + 255) / 256, 256, 0, stream>>>(act2, lw3, lb3, out);
    (void)in_sizes; (void)n_in; (void)out_size; (void)ws_size;
}
